// SAModule_68264210202827
// MI455X (gfx1250) — matrix operand order repaired and checked by reading, not compiled, not run on hardware
//
#include <hip/hip_runtime.h>
#include <math.h>

#define Bc    16
#define Nn    4096
#define CINc  64
#define COUTc 128
#define HIDc  128
#define Kk    32
#define Mm    1024
#define AKP   96     // padded K for GEMM1 (67 -> 96)
#define WPB   4      // waves per block in conv kernel

typedef _Float16 v16h __attribute__((ext_vector_type(16)));
typedef float    v8f  __attribute__((ext_vector_type(8)));

// ---------------------------------------------------------------- helpers ---

static __device__ inline v8f wmma_f16(v16h a, v16h b, v8f c) {
  // D(16x16,f32) = A(16x32,f16) * B(32x16,f16) + C
  return __builtin_amdgcn_wmma_f32_16x16x32_f16(
      false, a, false, b, (short)0, c, false, false);
}

// A fragment (16x32 f16) from a row-major LDS tile (ISA 7.12.2 layout).
static __device__ inline v16h frag_a_lds(const _Float16* A, int ld, int row0,
                                         int k0, int lane) {
  int r  = row0 + (lane & 15);
  int kb = k0 + ((lane & 16) ? 8 : 0);
  const _Float16* p = A + r * ld + kb;
  v16h a{};
#pragma unroll
  for (int e = 0; e < 8; ++e) { a[e] = p[e]; a[8 + e] = p[16 + e]; }
  return a;
}

// mish(v) = v*tanh(softplus(v)).  With p=e^v, u=p(p+2):
// tanh(log1p(p)) = u/(u+2)  ->  one v_exp_f32 + one v_rcp_f32 + a few muls.
static __device__ inline float mishf(float v) {
  float p = __expf(v);
  float u = p * (p + 2.f);
  float r = v * u * __builtin_amdgcn_rcpf(u + 2.f);
  return (v > 20.f) ? v : r;                 // large-v: tanh(sp)->1
}

// sorted 4-entry (val,idx) insert; ties -> lower index first
static __device__ inline void ins4(float d, int p,
                                   float& q0, float& q1, float& q2, float& q3,
                                   int& j0, int& j1, int& j2, int& j3) {
  if (d < q3 || (d == q3 && p < j3)) {
    if (d < q0 || (d == q0 && p < j0))      { q3=q2;j3=j2; q2=q1;j2=j1; q1=q0;j1=j0; q0=d;j0=p; }
    else if (d < q1 || (d == q1 && p < j1)) { q3=q2;j3=j2; q2=q1;j2=j1; q1=d;j1=p; }
    else if (d < q2 || (d == q2 && p < j2)) { q3=q2;j3=j2; q2=d;j2=p; }
    else                                    { q3=d;j3=p; }
  }
}

// ------------------------------------------------------------ weight pack ---
// Repack a row-major [kmax x 128] f32 weight into f16 B-fragment lane order.
// Rows >= kmax become 0 (K padding). One packed v16h == one lane's fragment.
__global__ __launch_bounds__(256) void pack_w_kernel(const float* __restrict__ W,
                                                     _Float16* __restrict__ dst,
                                                     int KT, int kmax) {
  int t = blockIdx.x * 256 + threadIdx.x;
  int total = KT * 8 * 32 * 16;
  if (t >= total) return;
  int e    = t & 15;
  int lane = (t >> 4) & 31;
  int tile = t >> 9;
  int kt = tile >> 3, nt = tile & 7;
  int col = nt * 16 + (lane & 15);
  int kk  = kt * 32 + ((lane & 16) ? 8 : 0) + (e & 7) + ((e & 8) ? 16 : 0);
  dst[t] = (kk < kmax) ? (_Float16)W[kk * COUTc + col] : (_Float16)0.f;
}

// -------------------------------------------------------------------- FPS ---
__global__ __launch_bounds__(1024) void fps_kernel(const float* __restrict__ pos,
                                                   int* __restrict__ idx,
                                                   float* __restrict__ pos_out) {
  const int b    = blockIdx.x;
  const int tid  = threadIdx.x;
  const int lane = tid & 31, wave = tid >> 5;
  __shared__ float s_val[32];
  __shared__ int   s_idx[32];
  __shared__ int   s_win;

  const float* pb = pos + (size_t)b * Nn * 3;
  float px[4], py[4], pz[4], mind[4];
#pragma unroll
  for (int j = 0; j < 4; ++j) {
    int p = tid * 4 + j;
    px[j] = pb[p * 3 + 0]; py[j] = pb[p * 3 + 1]; pz[j] = pb[p * 3 + 2];
    mind[j] = __builtin_huge_valf();
  }
  if (tid == 0) idx[b * Mm] = 0;
  float lx = pb[0], ly = pb[1], lz = pb[2];

  for (int m = 1; m < Mm; ++m) {
    float bestv = -1.0f; int besti = 0x7fffffff;
#pragma unroll
    for (int j = 0; j < 4; ++j) {
      float dx = px[j] - lx, dy = py[j] - ly, dz = pz[j] - lz;
      float d  = dx * dx + dy * dy + dz * dz;
      mind[j]  = fminf(mind[j], d);
      if (mind[j] > bestv) { bestv = mind[j]; besti = tid * 4 + j; }
    }
#pragma unroll
    for (int off = 16; off > 0; off >>= 1) {      // wave32 butterfly argmax
      float ov = __shfl_xor(bestv, off, 32);
      int   oi = __shfl_xor(besti, off, 32);
      if (ov > bestv || (ov == bestv && oi < besti)) { bestv = ov; besti = oi; }
    }
    if (lane == 0) { s_val[wave] = bestv; s_idx[wave] = besti; }
    __syncthreads();
    if (wave == 0) {
      bestv = s_val[lane]; besti = s_idx[lane];
#pragma unroll
      for (int off = 16; off > 0; off >>= 1) {
        float ov = __shfl_xor(bestv, off, 32);
        int   oi = __shfl_xor(besti, off, 32);
        if (ov > bestv || (ov == bestv && oi < besti)) { bestv = ov; besti = oi; }
      }
      if (lane == 0) { s_win = besti; idx[b * Mm + m] = besti; }
    }
    __syncthreads();
    int win = s_win;
    lx = pb[win * 3 + 0]; ly = pb[win * 3 + 1]; lz = pb[win * 3 + 2];
  }
  __syncthreads();
  for (int m = tid; m < Mm; m += 1024) {
    int i = idx[b * Mm + m];
    pos_out[(b * Mm + m) * 3 + 0] = pb[i * 3 + 0];
    pos_out[(b * Mm + m) * 3 + 1] = pb[i * 3 + 1];
    pos_out[(b * Mm + m) * 3 + 2] = pb[i * 3 + 2];
  }
}

// -------------------------------------------------------------------- KNN ---
// One wave per centroid. Single distance pass builds a per-lane sorted top-4
// register queue (lane owns points p == lane mod 32); each of the 32 picks is
// a wave32 butterfly argmin over queue heads; winning lane pops + marks LDS.
// Rare queue-drain rebuild rescans that lane's 128 LDS entries.
__global__ __launch_bounds__(64) void knn_kernel(const float* __restrict__ pos,
                                                 const int* __restrict__ idx,
                                                 int* __restrict__ nbr) {
  __shared__ float d2s[2][Nn];
  const int w = threadIdx.x >> 5, lane = threadIdx.x & 31;
  const int c = blockIdx.x * 2 + w;
  const int b = c >> 10;
  const float* pb = pos + (size_t)b * Nn * 3;
  const int ic = idx[c];
  const float qx = pb[ic * 3 + 0], qy = pb[ic * 3 + 1], qz = pb[ic * 3 + 2];
  const float INF = __builtin_huge_valf();

  float q0 = INF, q1 = INF, q2 = INF, q3 = INF;
  int   j0 = 0x7fffffff, j1 = 0x7fffffff, j2 = 0x7fffffff, j3 = 0x7fffffff;

  for (int p = lane; p < Nn; p += 32) {
    float dx = pb[p * 3 + 0] - qx, dy = pb[p * 3 + 1] - qy, dz = pb[p * 3 + 2] - qz;
    float d  = dx * dx + dy * dy + dz * dz;
    d2s[w][p] = d;
    ins4(d, p, q0, q1, q2, q3, j0, j1, j2, j3);
  }
  int nq = 4;
  __builtin_amdgcn_wave_barrier();

  for (int kk = 0; kk < Kk; ++kk) {
    float bv = q0; int bi = j0;
#pragma unroll
    for (int off = 16; off > 0; off >>= 1) {
      float ov = __shfl_xor(bv, off, 32);
      int   oi = __shfl_xor(bi, off, 32);
      if (ov < bv || (ov == bv && oi < bi)) { bv = ov; bi = oi; }
    }
    if (lane == 0) nbr[c * Kk + kk] = bi;
    if ((bi & 31) == lane) {                 // owner lane pops its head
      d2s[w][bi] = INF;                      // mark chosen for rebuilds
      q0 = q1; j0 = j1; q1 = q2; j1 = j2; q2 = q3; j2 = j3;
      q3 = INF; j3 = 0x7fffffff;
      if (--nq == 0) {                       // rare: rebuild from LDS
        q0 = q1 = q2 = q3 = INF;
        j0 = j1 = j2 = j3 = 0x7fffffff;
        for (int p = lane; p < Nn; p += 32) {
          float d = d2s[w][p];
          ins4(d, p, q0, q1, q2, q3, j0, j1, j2, j3);
        }
        nq = 4;
      }
    }
    __builtin_amdgcn_wave_barrier();
  }
}

// -------------------------------------------------- skip GEMM (WMMA) ---------
// out[c, :] = x[idx[c]] @ Wlin + blin   as a [16384x64]x[64x128] WMMA GEMM.
__global__ __launch_bounds__(64) void skip_kernel(const float* __restrict__ x,
                                                  const float* __restrict__ blin,
                                                  const _Float16* __restrict__ Wlp,
                                                  const int* __restrict__ idx,
                                                  float* __restrict__ out) {
  __shared__ _Float16 Ax[2][16][CINc];             // [16 x 64] f16 per wave
  const int w = threadIdx.x >> 5, lane = threadIdx.x & 31;
  const int tile = blockIdx.x * 2 + w;             // 1024 tiles

  {
    const int row = lane & 15;
    const int c   = tile * 16 + row;
    const int b   = c >> 10;
    const int ic  = idx[c];
    const float* xr = x + ((size_t)b * Nn + ic) * CINc + (lane >> 4) * 32;
    _Float16* dst   = &Ax[w][row][(lane >> 4) * 32];
#pragma unroll 8
    for (int f = 0; f < 32; ++f) dst[f] = (_Float16)xr[f];
  }
  __builtin_amdgcn_wave_barrier();

  v16h a0 = frag_a_lds(&Ax[w][0][0], CINc, 0, 0,  lane);
  v16h a1 = frag_a_lds(&Ax[w][0][0], CINc, 0, 32, lane);
  const v16h* Wv = (const v16h*)Wlp;

  const int colL = lane & 15;
  const int rofs = (lane & 16) ? 8 : 0;
#pragma unroll
  for (int nt = 0; nt < 8; ++nt) {
    v8f acc = {};
    acc = wmma_f16(a0, Wv[(0 * 8 + nt) * 32 + lane], acc);
    acc = wmma_f16(a1, Wv[(1 * 8 + nt) * 32 + lane], acc);
    const int   col = nt * 16 + colL;
    const float bb  = blin[col];
#pragma unroll
    for (int r = 0; r < 8; ++r)
      out[(size_t)(tile * 16 + r + rofs) * COUTc + col] = acc[r] + bb;
  }
}

// --------------------------------------------------- fused gather+MLP+max ---
__global__ __launch_bounds__(32 * WPB) void conv_kernel(
    const float* __restrict__ x,    const float* __restrict__ pos,
    const _Float16* __restrict__ W1p, const float* __restrict__ b1,
    const _Float16* __restrict__ W2p, const float* __restrict__ b2,
    const int* __restrict__ idx,    const int* __restrict__ nbr,
    float* __restrict__ out) {
  __shared__ _Float16 As[WPB][Kk][AKP];      // gathered [32 x 96] f16 (24 KB)
  __shared__ _Float16 Hs[WPB][Kk][HIDc];     // mish(h@W1+b1) as f16  (32 KB)

  const int w = threadIdx.x >> 5, lane = threadIdx.x & 31;
  const int c = blockIdx.x * WPB + w;
  const int b = c >> 10;
  const float* pb = pos + (size_t)b * Nn * 3;
  const float* xb = x   + (size_t)b * Nn * CINc;
  const int ic = idx[c];

  __builtin_prefetch(W1p, 0, 3);
  __builtin_prefetch(W2p, 0, 3);

  // ---- stage: lane k builds row k of A = [x[nbr], rel, 0-pad] in f16 ----
  {
    const int  nk = nbr[c * Kk + lane];
    const float qx = pb[ic * 3 + 0], qy = pb[ic * 3 + 1], qz = pb[ic * 3 + 2];
    const float* xr = xb + (size_t)nk * CINc;
#pragma unroll 8
    for (int f = 0; f < CINc; ++f) As[w][lane][f] = (_Float16)xr[f];
    As[w][lane][64] = (_Float16)(pb[nk * 3 + 0] - qx);
    As[w][lane][65] = (_Float16)(pb[nk * 3 + 1] - qy);
    As[w][lane][66] = (_Float16)(pb[nk * 3 + 2] - qz);
#pragma unroll
    for (int f = 67; f < AKP; ++f) As[w][lane][f] = (_Float16)0.f;
  }
  __builtin_amdgcn_wave_barrier();

  const int colL = lane & 15;
  const int rofs = (lane & 16) ? 8 : 0;
  const v16h* W1v = (const v16h*)W1p;
  const v16h* W2v = (const v16h*)W2p;

  // ---- GEMM1: [32x96]x[96x128] (+b1, mish) -> Hs ----
  v16h a1[2][3];
#pragma unroll
  for (int mt = 0; mt < 2; ++mt)
#pragma unroll
    for (int kt = 0; kt < 3; ++kt)
      a1[mt][kt] = frag_a_lds(&As[w][0][0], AKP, mt * 16, kt * 32, lane);

#pragma unroll
  for (int nt = 0; nt < 8; ++nt) {
    v8f acc0 = {}, acc1 = {};
#pragma unroll
    for (int kt = 0; kt < 3; ++kt) {
      v16h bf = W1v[(kt * 8 + nt) * 32 + lane];
      acc0 = wmma_f16(a1[0][kt], bf, acc0);
      acc1 = wmma_f16(a1[1][kt], bf, acc1);
    }
    const int   col  = nt * 16 + colL;
    const float bias = b1[col];
#pragma unroll
    for (int r = 0; r < 8; ++r) {
      Hs[w][r + rofs][col]      = (_Float16)mishf(acc0[r] + bias);
      Hs[w][16 + r + rofs][col] = (_Float16)mishf(acc1[r] + bias);
    }
  }
  __builtin_amdgcn_wave_barrier();

  // ---- GEMM2: [32x128]x[128x128], then max over the 32 neighbor rows ----
  v16h a2[2][4];
#pragma unroll
  for (int mt = 0; mt < 2; ++mt)
#pragma unroll
    for (int kt = 0; kt < 4; ++kt)
      a2[mt][kt] = frag_a_lds(&Hs[w][0][0], HIDc, mt * 16, kt * 32, lane);

#pragma unroll
  for (int nt = 0; nt < 8; ++nt) {
    v8f acc0 = {}, acc1 = {};
#pragma unroll
    for (int kt = 0; kt < 4; ++kt) {
      v16h bf = W2v[(kt * 8 + nt) * 32 + lane];
      acc0 = wmma_f16(a2[0][kt], bf, acc0);
      acc1 = wmma_f16(a2[1][kt], bf, acc1);
    }
    const int col = nt * 16 + colL;
    // lanes 0-15 hold rows {0-7,16-23}, lanes 16-31 rows {8-15,24-31}
    float mx = -__builtin_huge_valf();
#pragma unroll
    for (int r = 0; r < 8; ++r) mx = fmaxf(mx, fmaxf(acc0[r], acc1[r]));
    mx = fmaxf(mx, __shfl_xor(mx, 16, 32));
    mx += b2[col];                       // max(h + b2) == max(h) + b2
    if (lane < 16) {
      size_t o = (size_t)c * COUTc + col;
      out[o] = out[o] + mx;              // skip kernel wrote x[idx]@Wlin+blin
    }
  }
}

// ------------------------------------------------------------------ launch ---
extern "C" void kernel_launch(void* const* d_in, const int* in_sizes, int n_in,
                              void* d_out, int out_size, void* d_ws, size_t ws_size,
                              hipStream_t stream) {
  (void)in_sizes; (void)n_in; (void)out_size; (void)ws_size;
  const float* x    = (const float*)d_in[0];
  const float* pos  = (const float*)d_in[1];
  const float* W1   = (const float*)d_in[2];
  const float* b1   = (const float*)d_in[3];
  const float* W2   = (const float*)d_in[4];
  const float* b2   = (const float*)d_in[5];
  const float* Wlin = (const float*)d_in[6];
  const float* blin = (const float*)d_in[7];

  float* out     = (float*)d_out;
  float* pos_out = out + (size_t)Bc * Mm * COUTc;

  // workspace layout
  int* idx = (int*)d_ws;                              // 16384 ints
  int* nbr = idx + Bc * Mm;                           // 16384*32 ints
  _Float16* W1p = (_Float16*)(nbr + Bc * Mm * Kk);    // 3*8*32*16 = 12288 halfs
  _Float16* W2p = W1p + 3 * 8 * 32 * 16;              // 4*8*32*16 = 16384 halfs
  _Float16* Wlp = W2p + 4 * 8 * 32 * 16;              // 2*8*32*16 =  8192 halfs

  pack_w_kernel<<<48, 256, 0, stream>>>(W1,   W1p, 3, CINc + 3);
  pack_w_kernel<<<64, 256, 0, stream>>>(W2,   W2p, 4, HIDc);
  pack_w_kernel<<<32, 256, 0, stream>>>(Wlin, Wlp, 2, CINc);

  fps_kernel<<<Bc, 1024, 0, stream>>>(pos, idx, pos_out);
  knn_kernel<<<(Bc * Mm) / 2, 64, 0, stream>>>(pos, idx, nbr);
  skip_kernel<<<(Bc * Mm) / 32, 64, 0, stream>>>(x, blin, Wlp, idx, out);
  conv_kernel<<<(Bc * Mm) / WPB, 32 * WPB, 0, stream>>>(
      x, pos, W1p, b1, W2p, b2, idx, nbr, out);
}
